// AdditiveSurrogate_28948079575711
// MI455X (gfx1250) — matrix operand order repaired and checked by reading, not compiled, not run on hardware
//
#include <hip/hip_runtime.h>

// CDNA5 / gfx1250: wave32, WMMA 16x16x32 f16 -> f32 accum.
typedef __attribute__((ext_vector_type(16))) _Float16 v16h;
typedef __attribute__((ext_vector_type(8)))  float    v8f;

#define B_ROWS 16384
#define F_DIM  256
#define H_DIM  64
#define NFG    8                         // number of f-groups (grid.y)
#define F_PER_G (F_DIM / NFG)            // 32 f per group
#define CHUNK_HALFS 512                  // 32 lanes * 16 halfs per chunk
#define HALFS_PER_F (6 * CHUNK_HALFS)    // 6 chunks: W1a,b1a,W1b,b1b,W2(B0),W2(B1)
#define IMG_BYTES ((size_t)F_DIM * HALFS_PER_F * 2)   // 1.5 MB f16 weight image

// ---------------------------------------------------------------------------
// Prep: swizzle W1/b1/W2 (f32, [256,64]) into the exact per-lane f16 layout the
// WMMA A/B operands need, so the hot loop is pure coalesced b128 loads.
//   A 16x32 f16 layout: lane<16 holds K {0..7,16..23}; lane>=16 holds {8..15,24..31}
//   B 32x16 f16 layout (broadcast over N): same K order as A: lane<16 holds K {0..7,16..23}; lane>=16 holds {8..15,24..31}
// ---------------------------------------------------------------------------
__global__ void addsur_prep_weights(const float* __restrict__ W1,
                                    const float* __restrict__ b1,
                                    const float* __restrict__ W2,
                                    _Float16* __restrict__ img) {
    const int f = blockIdx.x;          // 0..255
    const int l = threadIdx.x;         // 0..31
    const int half = l >> 4;
    _Float16* dst = img + (size_t)f * HALFS_PER_F + l * 16;
    const float* w1f = W1 + f * H_DIM;
    const float* b1f = b1 + f * H_DIM;
    const float* w2f = W2 + f * H_DIM;
    for (int j = 0; j < 16; ++j) {
        const int K  = (j < 8) ? (half * 8 + j) : (16 + half * 8 + (j - 8));
        const int Kb = K;
        dst[0 * CHUNK_HALFS + j] = (_Float16)w1f[K];        // A0: W1, h = K
        dst[1 * CHUNK_HALFS + j] = (_Float16)b1f[K];        // A0: b1
        dst[2 * CHUNK_HALFS + j] = (_Float16)w1f[32 + K];   // A1: W1, h = 32+K
        dst[3 * CHUNK_HALFS + j] = (_Float16)b1f[32 + K];   // A1: b1
        dst[4 * CHUNK_HALFS + j] = (_Float16)w2f[Kb];       // B0: W2, h = Kb (bcast over N)
        dst[5 * CHUNK_HALFS + j] = (_Float16)w2f[32 + Kb];  // B1: W2, h = 32+Kb
    }
}

// Deterministic LDS tree-reduction of b2 (256 values -> scalar).
__global__ void addsur_prep_b2(const float* __restrict__ b2, float* __restrict__ out) {
    __shared__ float s[256];
    const int t = threadIdx.x;
    s[t] = b2[t];
    __syncthreads();
    for (int k = 128; k > 0; k >>= 1) {
        if (t < k) s[t] += s[t + k];
        __syncthreads();
    }
    if (t == 0) *out = s[0];
}

// ---------------------------------------------------------------------------
// Main: each wave owns one 16-row batch tile and one f-group of 32 features.
// Per f: build A = relu(x*W1+b1) with packed-f16 VALU, then two
// v_wmma_f32_16x16x32_f16 accumulate the W2 contraction (h=0..31, 32..63)
// into a persistent f32 accumulator. After the loop, D's column 0 holds the
// per-row partial sum for this f-group (B columns are identical by design).
// ---------------------------------------------------------------------------
__global__ void __launch_bounds__(256)
addsur_main(const float* __restrict__ x,
            const _Float16* __restrict__ img,
            float* __restrict__ partial) {
    const int wave = threadIdx.x >> 5;
    const int lane = threadIdx.x & 31;
    const int b0   = (blockIdx.x * 8 + wave) * 16;   // batch tile base row
    const int row  = lane & 15;                      // A-matrix row for this lane
    const long xbase = (long)(b0 + row) * F_DIM;
    const int f0 = blockIdx.y * F_PER_G;

    const _Float16* p = img + (size_t)f0 * HALFS_PER_F + lane * 16;

    v8f  acc = {};
    v16h zz  = {};

    for (int i = 0; i < F_PER_G; ++i) {
        const _Float16* pw = p + (size_t)i * HALFS_PER_F;
        // per-lane activation scalar x[b_row, f]
        const float s = x[xbase + f0 + i];
        const _Float16 sh = (_Float16)s;
        const v16h sv = zz + sh;                     // splat

        const v16h w1a = *(const v16h*)(pw + 0 * CHUNK_HALFS);
        const v16h c1a = *(const v16h*)(pw + 1 * CHUNK_HALFS);
        const v16h w1b = *(const v16h*)(pw + 2 * CHUNK_HALFS);
        const v16h c1b = *(const v16h*)(pw + 3 * CHUNK_HALFS);
        const v16h bb0 = *(const v16h*)(pw + 4 * CHUNK_HALFS);
        const v16h bb1 = *(const v16h*)(pw + 5 * CHUNK_HALFS);
        __builtin_prefetch(pw + HALFS_PER_F, 0, 1);  // global_prefetch_b8 next f

        // relu(x*W1 + b1) in packed f16 (v_pk_fma_f16 / v_pk_max)
        const v16h a0 = __builtin_elementwise_max(__builtin_elementwise_fma(sv, w1a, c1a), zz);
        const v16h a1 = __builtin_elementwise_max(__builtin_elementwise_fma(sv, w1b, c1b), zz);

        // D += A0 * B0 ; D += A1 * B1   (f32 accumulate on the matrix pipe)
        acc = __builtin_amdgcn_wmma_f32_16x16x32_f16(false, a0, false, bb0,
                                                     (short)0, acc, false, false);
        acc = __builtin_amdgcn_wmma_f32_16x16x32_f16(false, a1, false, bb1,
                                                     (short)0, acc, false, false);
    }

    // C/D layout: VGPR r, lanes 0-15 -> M=r, lanes 16-31 -> M=r+8; N = lane%16.
    // All N columns are identical (B broadcast), so read column 0 from lanes 0 & 16.
    if ((lane & 15) == 0) {
        float* pp = partial + (size_t)blockIdx.y * B_ROWS + b0 + (lane >> 4) * 8;
#pragma unroll
        for (int r = 0; r < 8; ++r) pp[r] = acc[r];
    }
}

// out[b] = sum over f-groups + sum(b2)
__global__ void addsur_reduce(const float* __restrict__ partial,
                              const float* __restrict__ b2sum,
                              float* __restrict__ out) {
    const int b = blockIdx.x * blockDim.x + threadIdx.x;
    if (b >= B_ROWS) return;
    float a = *b2sum;
#pragma unroll
    for (int g = 0; g < NFG; ++g) a += partial[(size_t)g * B_ROWS + b];
    out[b] = a;
}

extern "C" void kernel_launch(void* const* d_in, const int* in_sizes, int n_in,
                              void* d_out, int out_size, void* d_ws, size_t ws_size,
                              hipStream_t stream) {
    const float* x  = (const float*)d_in[0];
    const float* W1 = (const float*)d_in[1];
    const float* b1 = (const float*)d_in[2];
    const float* W2 = (const float*)d_in[3];
    const float* b2 = (const float*)d_in[4];
    float* out = (float*)d_out;

    // Workspace layout (≈2.1 MB total):
    //   [0, IMG_BYTES)        f16 weight image
    //   [IMG_BYTES, +256)     b2 scalar sum (padded)
    //   [+256, +NFG*B*4)      per-f-group partial row sums
    char* ws = (char*)d_ws;
    _Float16* img   = (_Float16*)ws;
    float* b2sum    = (float*)(ws + IMG_BYTES);
    float* partials = (float*)(ws + IMG_BYTES + 256);

    addsur_prep_weights<<<F_DIM, 32, 0, stream>>>(W1, b1, W2, img);
    addsur_prep_b2<<<1, 256, 0, stream>>>(b2, b2sum);
    addsur_main<<<dim3(B_ROWS / 128, NFG), 256, 0, stream>>>(x, img, partials);
    addsur_reduce<<<B_ROWS / 256, 256, 0, stream>>>(partials, b2sum, out);
}
